// FlatQuantQwen3Attention_70875550318900
// MI455X (gfx1250) — 8-bit matrix operand order repaired and checked by reading, compiled, not run on hardware
//
#include <hip/hip_runtime.h>
#include <cstdint>

#define B_     2
#define S_     2048
#define H_     2048
#define HV_    32
#define HK_    16
#define DK_    128
#define DV_    128
#define KDIM_  2048
#define VDIM_  4096
#define L_     32
#define R_     64
#define NPROJ_ 12352
#define MIXW_  8192
#define NTOK_  (B_ * S_)

#define OFF_Q  0
#define OFF_K  2048
#define OFF_V  4096
#define OFF_Z  8192
#define OFF_B  12288
#define OFF_A  12320

typedef __attribute__((ext_vector_type(8))) int v8i;

#if defined(__gfx1250__) && __has_builtin(__builtin_amdgcn_global_load_async_to_lds_b128)
#define HAVE_ASYNC_LDS 1
typedef int async_v4i __attribute__((vector_size(16)));
#define ASYNC_GSRC(p) ((__attribute__((address_space(1))) async_v4i*)(p))
#define ASYNC_LDST(p) ((__attribute__((address_space(3))) async_v4i*)(p))
#if __has_builtin(__builtin_amdgcn_s_wait_asynccnt)
#define WAIT_ASYNC0() __builtin_amdgcn_s_wait_asynccnt(0)
#else
#define WAIT_ASYNC0() asm volatile("s_wait_asynccnt 0" ::: "memory")
#endif
#endif

__global__ void k_dtrans_fq_act(const float* __restrict__ x,
                                const float* __restrict__ Pl,
                                const float* __restrict__ Pr,
                                int8_t* __restrict__ aq,
                                float* __restrict__ ascale) {
  __shared__ float X[L_ * R_];
  __shared__ float Y1[L_ * R_];
  __shared__ float red[256];
  const int tok = blockIdx.x;
  const float* xr = x + (size_t)tok * H_;
  for (int i = threadIdx.x; i < L_ * R_; i += 256) X[i] = xr[i];
  __syncthreads();
  for (int i = threadIdx.x; i < L_ * R_; i += 256) {
    int m = i / R_, r = i % R_;
    float acc = 0.f;
    for (int l = 0; l < L_; ++l) acc += Pl[l * L_ + m] * X[l * R_ + r];
    Y1[i] = acc;
  }
  __syncthreads();
  float yv[8];
  float lmax = 0.f;
  for (int ii = 0; ii < 8; ++ii) {
    int i = threadIdx.x + ii * 256;
    int m = i / R_, n = i % R_;
    float acc = 0.f;
    for (int r = 0; r < R_; ++r) acc += Y1[m * R_ + r] * Pr[r * R_ + n];
    yv[ii] = acc;
    lmax = fmaxf(lmax, fabsf(acc));
  }
  red[threadIdx.x] = lmax;
  __syncthreads();
  for (int s = 128; s > 0; s >>= 1) {
    if (threadIdx.x < s) red[threadIdx.x] = fmaxf(red[threadIdx.x], red[threadIdx.x + s]);
    __syncthreads();
  }
  const float sc = fmaxf(red[0] / 127.f, 1e-8f);
  const float inv = 1.f / sc;
  if (threadIdx.x == 0) ascale[tok] = sc;
  int8_t* dst = aq + (size_t)tok * H_;
  for (int ii = 0; ii < 8; ++ii) {
    int i = threadIdx.x + ii * 256;
    float q = fminf(fmaxf(rintf(yv[ii] * inv), -128.f), 127.f);
    dst[i] = (int8_t)q;
  }
}

__global__ void k_dtrans_fq_rows(const float* __restrict__ W,
                                 const float* __restrict__ Pl,
                                 const float* __restrict__ Pr,
                                 int8_t* __restrict__ Wall,
                                 float* __restrict__ wscale,
                                 int rowoff) {
  __shared__ float X[L_ * R_];
  __shared__ float Y1[L_ * R_];
  __shared__ float red[256];
  const int row = blockIdx.x;
  const float* xr = W + (size_t)row * H_;
  for (int i = threadIdx.x; i < L_ * R_; i += 256) X[i] = xr[i];
  __syncthreads();
  for (int i = threadIdx.x; i < L_ * R_; i += 256) {
    int m = i / R_, r = i % R_;
    float acc = 0.f;
    for (int l = 0; l < L_; ++l) acc += Pl[l * L_ + m] * X[l * R_ + r];
    Y1[i] = acc;
  }
  __syncthreads();
  float yv[8];
  float lmax = 0.f;
  for (int ii = 0; ii < 8; ++ii) {
    int i = threadIdx.x + ii * 256;
    int m = i / R_, n = i % R_;
    float acc = 0.f;
    for (int r = 0; r < R_; ++r) acc += Y1[m * R_ + r] * Pr[r * R_ + n];
    yv[ii] = acc;
    lmax = fmaxf(lmax, fabsf(acc));
  }
  red[threadIdx.x] = lmax;
  __syncthreads();
  for (int s = 128; s > 0; s >>= 1) {
    if (threadIdx.x < s) red[threadIdx.x] = fmaxf(red[threadIdx.x], red[threadIdx.x + s]);
    __syncthreads();
  }
  const float sc = fmaxf(red[0] / 127.f, 1e-8f);
  const float inv = 1.f / sc;
  if (threadIdx.x == 0) wscale[rowoff + row] = sc;
  int8_t* dst = Wall + (size_t)(rowoff + row) * H_;
  for (int ii = 0; ii < 8; ++ii) {
    int i = threadIdx.x + ii * 256;
    float q = fminf(fmaxf(rintf(yv[ii] * inv), -128.f), 127.f);
    dst[i] = (int8_t)q;
  }
}

__global__ void k_dtrans_rows_f32(const float* __restrict__ W,
                                  const float* __restrict__ Pl,
                                  const float* __restrict__ Pr,
                                  float* __restrict__ out) {
  __shared__ float X[L_ * R_];
  __shared__ float Y1[L_ * R_];
  const int row = blockIdx.x;
  const float* xr = W + (size_t)row * H_;
  for (int i = threadIdx.x; i < L_ * R_; i += 256) X[i] = xr[i];
  __syncthreads();
  for (int i = threadIdx.x; i < L_ * R_; i += 256) {
    int m = i / R_, r = i % R_;
    float acc = 0.f;
    for (int l = 0; l < L_; ++l) acc += Pl[l * L_ + m] * X[l * R_ + r];
    Y1[i] = acc;
  }
  __syncthreads();
  float* dst = out + (size_t)row * H_;
  for (int i = threadIdx.x; i < L_ * R_; i += 256) {
    int m = i / R_, n = i % R_;
    float acc = 0.f;
    for (int r = 0; r < R_; ++r) acc += Y1[m * R_ + r] * Pr[r * R_ + n];
    dst[i] = acc;
  }
}

__global__ void k_vt_mix_fq(const float* __restrict__ Wvt,
                            const float* __restrict__ Vt,
                            int8_t* __restrict__ Wall,
                            float* __restrict__ wscale,
                            int rowoff) {
  __shared__ float red[256];
  const int row = blockIdx.x;
  const int h = row / DV_;
  const int e = row % DV_;
  float yv[8];
  float lmax = 0.f;
  for (int ii = 0; ii < 8; ++ii) {
    int i = threadIdx.x + ii * 256;
    float acc = 0.f;
    for (int d = 0; d < DV_; ++d)
      acc += Wvt[((size_t)(h * DV_ + d)) * H_ + i] * Vt[d * DV_ + e];
    yv[ii] = acc;
    lmax = fmaxf(lmax, fabsf(acc));
  }
  red[threadIdx.x] = lmax;
  __syncthreads();
  for (int s = 128; s > 0; s >>= 1) {
    if (threadIdx.x < s) red[threadIdx.x] = fmaxf(red[threadIdx.x], red[threadIdx.x + s]);
    __syncthreads();
  }
  const float sc = fmaxf(red[0] / 127.f, 1e-8f);
  const float inv = 1.f / sc;
  if (threadIdx.x == 0) wscale[rowoff + row] = sc;
  int8_t* dst = Wall + (size_t)(rowoff + row) * H_;
  for (int ii = 0; ii < 8; ++ii) {
    int i = threadIdx.x + ii * 256;
    float q = fminf(fmaxf(rintf(yv[ii] * inv), -128.f), 127.f);
    dst[i] = (int8_t)q;
  }
}

__global__ void k_weff_stage1(const float* __restrict__ Wo,
                              const float* __restrict__ Ot,
                              float* __restrict__ T1) {
  const int o = blockIdx.x;
  for (int ii = 0; ii < 16; ++ii) {
    int i = threadIdx.x + ii * 256;
    int g = i >> 7, d = i & 127;
    float acc = 0.f;
    for (int h = 0; h < HV_; ++h)
      acc += Wo[(size_t)o * VDIM_ + h * DV_ + d] * Ot[h * HV_ + g];
    T1[(size_t)o * VDIM_ + i] = acc;
  }
}

__global__ void k_weff_stage2_fq(const float* __restrict__ T1,
                                 const float* __restrict__ Vt,
                                 int8_t* __restrict__ Weff,
                                 float* __restrict__ wsc) {
  __shared__ float rowb[VDIM_];
  __shared__ float red[256];
  const int o = blockIdx.x;
  for (int i = threadIdx.x; i < VDIM_; i += 256) rowb[i] = T1[(size_t)o * VDIM_ + i];
  __syncthreads();
  float yv[16];
  float lmax = 0.f;
  for (int ii = 0; ii < 16; ++ii) {
    int i = threadIdx.x + ii * 256;
    int g = i >> 7, e = i & 127;
    float acc = 0.f;
    for (int d = 0; d < DV_; ++d) acc += rowb[g * DV_ + d] * Vt[d * DV_ + e];
    yv[ii] = acc;
    lmax = fmaxf(lmax, fabsf(acc));
  }
  red[threadIdx.x] = lmax;
  __syncthreads();
  for (int s = 128; s > 0; s >>= 1) {
    if (threadIdx.x < s) red[threadIdx.x] = fmaxf(red[threadIdx.x], red[threadIdx.x + s]);
    __syncthreads();
  }
  const float sc = fmaxf(red[0] / 127.f, 1e-8f);
  const float inv = 1.f / sc;
  if (threadIdx.x == 0) wsc[o] = sc;
  int8_t* dst = Weff + (size_t)o * VDIM_;
  for (int ii = 0; ii < 16; ++ii) {
    int i = threadIdx.x + ii * 256;
    float q = fminf(fmaxf(rintf(yv[ii] * inv), -128.f), 127.f);
    dst[i] = (int8_t)q;
  }
}

__global__ void __launch_bounds__(256)
k_gemm_i8_wmma(const int8_t* __restrict__ A, const float* __restrict__ sa,
               const int8_t* __restrict__ Bw, const float* __restrict__ sb,
               float* __restrict__ C, int M, int N, int K) {
  const int lane  = threadIdx.x & 31;
  const int wave  = threadIdx.x >> 5;
  const int ntile = blockIdx.x * 8 + wave;
  const int mtile = blockIdx.y;
  const int half  = lane >> 4;
#if defined(HAVE_ASYNC_LDS)
  const bool active = (ntile * 16) < N;
  const int ncol = (active ? ntile * 16 : 0) + (lane & 15);
  const int8_t* brow = Bw + (size_t)ncol * K;
  __shared__ __align__(16) int8_t atile[2][16 * 64];
  const int t = threadIdx.x;
  const int crow = t >> 2;
  const int cch  = (t & 3) << 4;
  const int8_t* agbase = A + (size_t)(mtile * 16 + crow) * K + cch;
  if (t < 64) {
    __builtin_amdgcn_global_load_async_to_lds_b128(
        ASYNC_GSRC(agbase), ASYNC_LDST(&atile[0][crow * 64 + cch]), 0, 0);
    WAIT_ASYNC0();
  }
  __syncthreads();
  v8i acc = {0, 0, 0, 0, 0, 0, 0, 0};
  const int nsteps = K >> 6;
  for (int ks = 0; ks < nsteps; ++ks) {
    const int cur = ks & 1;
    if ((ks + 1) < nsteps && t < 64) {
      __builtin_amdgcn_global_load_async_to_lds_b128(
          ASYNC_GSRC(agbase + (size_t)(ks + 1) * 64),
          ASYNC_LDST(&atile[cur ^ 1][crow * 64 + cch]), 0, 0);
    }
    const int8_t* al = &atile[cur][(lane & 15) * 64];
    const int k0 = ks * 64;
    v8i av, bv;
#pragma unroll
    for (int v = 0; v < 8; ++v) {
      const int ka = ((v >> 1) << 4) + ((v & 1) << 2) + (half << 3);
      av[v] = *(const int*)(al + ka);
      const int kb = k0 + ((v >> 1) << 4) + ((v & 1) << 2) + (half << 3);
      bv[v] = *(const int*)(brow + kb);
    }
    __builtin_prefetch(brow + k0 + 128, 0, 0);
    acc = __builtin_amdgcn_wmma_i32_16x16x64_iu8(
         true, av,  true, bv, acc,
         false,  false);
    if (t < 64) WAIT_ASYNC0();
    __syncthreads();
  }
  if (active) {
    const float sn = sb[ncol];
#pragma unroll
    for (int r = 0; r < 8; ++r) {
      const int m = mtile * 16 + half * 8 + r;
      C[(size_t)m * N + ncol] = sa[m] * sn * (float)acc[r];
    }
  }
#else
  if (ntile * 16 >= N) return;
  const int mrow = mtile * 16 + (lane & 15);
  const int ncol = ntile * 16 + (lane & 15);
  const int8_t* arow = A + (size_t)mrow * K;
  const int8_t* brow = Bw + (size_t)ncol * K;
  v8i acc = {0, 0, 0, 0, 0, 0, 0, 0};
  for (int k0 = 0; k0 < K; k0 += 64) {
    v8i av, bv;
#pragma unroll
    for (int v = 0; v < 8; ++v) {
      const int ka = k0 + ((v >> 1) << 4) + ((v & 1) << 2) + (half << 3);
      av[v] = *(const int*)(arow + ka);
      const int kb = k0 + ((v >> 1) << 4) + ((v & 1) << 2) + (half << 3);
      bv[v] = *(const int*)(brow + kb);
    }
    __builtin_prefetch(arow + k0 + 128, 0, 0);
    __builtin_prefetch(brow + k0 + 128, 0, 0);
    acc = __builtin_amdgcn_wmma_i32_16x16x64_iu8(true, av, true, bv, acc, false, false);
  }
  const float sn = sb[ncol];
#pragma unroll
  for (int r = 0; r < 8; ++r) {
    const int m = mtile * 16 + half * 8 + r;
    C[(size_t)m * N + ncol] = sa[m] * sn * (float)acc[r];
  }
#endif
}

__global__ void k_conv_silu(const float* __restrict__ qkvzba,
                            const float* __restrict__ cw,
                            const float* __restrict__ cb,
                            float* __restrict__ mixed) {
  const int tok = blockIdx.x;
  const int b = tok / S_;
  const int s = tok % S_;
  for (int c = threadIdx.x; c < MIXW_; c += 256) {
    float acc = cb[c];
#pragma unroll
    for (int i = 0; i < 4; ++i) {
      const int sp = s - 3 + i;
      const float xv = (sp >= 0) ? qkvzba[((size_t)(b * S_ + sp)) * NPROJ_ + c] : 0.f;
      acc += xv * cw[c * 4 + i];
    }
    const float sig = 1.f / (1.f + __expf(-acc));
    mixed[(size_t)tok * MIXW_ + c] = acc * sig;
  }
}

__global__ void k_qk_prep(const float* __restrict__ mixed,
                          const float* __restrict__ Kt,
                          float* __restrict__ qn, float* __restrict__ kn) {
  const int idx = blockIdx.x;
  const int hk = idx % HK_;
  const int tok = idx / HK_;
  const int e = threadIdx.x;
  __shared__ float qrow[DK_];
  __shared__ float krow[DK_];
  __shared__ float red[DK_];
  qrow[e] = mixed[(size_t)tok * MIXW_ + hk * DK_ + e];
  krow[e] = mixed[(size_t)tok * MIXW_ + KDIM_ + hk * DK_ + e];
  __syncthreads();
  float qe = 0.f, ke = 0.f;
  for (int d = 0; d < DK_; ++d) {
    const float kt = Kt[d * DK_ + e];
    qe += qrow[d] * kt;
    ke += krow[d] * kt;
  }
  red[e] = fabsf(qe); __syncthreads();
  for (int s = 64; s > 0; s >>= 1) { if (e < s) red[e] = fmaxf(red[e], red[e + s]); __syncthreads(); }
  float sc = fmaxf(red[0] / 127.f, 1e-8f);
  __syncthreads();
  const float qq = fminf(fmaxf(rintf(qe / sc), -128.f), 127.f) * sc;
  red[e] = fabsf(ke); __syncthreads();
  for (int s = 64; s > 0; s >>= 1) { if (e < s) red[e] = fmaxf(red[e], red[e + s]); __syncthreads(); }
  sc = fmaxf(red[0] / 127.f, 1e-8f);
  __syncthreads();
  const float kq = fminf(fmaxf(rintf(ke / sc), -128.f), 127.f) * sc;
  red[e] = qq * qq; __syncthreads();
  for (int s = 64; s > 0; s >>= 1) { if (e < s) red[e] += red[e + s]; __syncthreads(); }
  const float qr = rsqrtf(red[0] + 1e-6f);
  __syncthreads();
  red[e] = kq * kq; __syncthreads();
  for (int s = 64; s > 0; s >>= 1) { if (e < s) red[e] += red[e + s]; __syncthreads(); }
  const float kr = rsqrtf(red[0] + 1e-6f);
  const size_t ob = ((size_t)tok * HV_ + hk * 2) * DK_;
  qn[ob + e] = qq * qr;  qn[ob + DK_ + e] = qq * qr;
  kn[ob + e] = kq * kr;  kn[ob + DK_ + e] = kq * kr;
}

__global__ void k_v_prep(const float* __restrict__ mixed, float* __restrict__ vf) {
  const int idx = blockIdx.x;
  const int hv = idx % HV_;
  const int tok = idx / HV_;
  const int d = threadIdx.x;
  __shared__ float red[DV_];
  const float x = mixed[(size_t)tok * MIXW_ + 2 * KDIM_ + hv * DV_ + d];
  red[d] = fabsf(x); __syncthreads();
  for (int s = 64; s > 0; s >>= 1) { if (d < s) red[d] = fmaxf(red[d], red[d + s]); __syncthreads(); }
  const float sc = fmaxf(red[0] / 127.f, 1e-8f);
  vf[((size_t)tok * HV_ + hv) * DV_ + d] =
      fminf(fmaxf(rintf(x / sc), -128.f), 127.f) * sc;
}

__global__ void k_beta_g(const float* __restrict__ qkvzba,
                         const float* __restrict__ A_log,
                         const float* __restrict__ dt_bias,
                         float* __restrict__ beta, float* __restrict__ g) {
  const int i = blockIdx.x * 256 + threadIdx.x;
  const int h = i % HV_;
  const int tok = i / HV_;
  const float bv = qkvzba[(size_t)tok * NPROJ_ + OFF_B + h];
  const float av = qkvzba[(size_t)tok * NPROJ_ + OFF_A + h];
  beta[i] = 1.f / (1.f + __expf(-bv));
  const float x = av + dt_bias[h];
  const float sp = (x > 20.f) ? x : log1pf(__expf(x));
  g[i] = -__expf(A_log[h]) * sp;
}

__global__ void __launch_bounds__(256)
k_delta_scan(const float* __restrict__ qn, const float* __restrict__ kn,
             const float* __restrict__ vf, const float* __restrict__ g,
             const float* __restrict__ beta, float* __restrict__ core) {
  const int b = blockIdx.x / HV_;
  const int h = blockIdx.x % HV_;
  const int t = threadIdx.x;
  const int v = t & 127;
  const int kh = t >> 7;
  __shared__ float lk[DK_];
  __shared__ float lq[DK_];
  __shared__ float red[256];
  float Sreg[64];
#pragma unroll
  for (int j = 0; j < 64; ++j) Sreg[j] = 0.f;
  const float qscale = 0.08838834764831845f;

  for (int s = 0; s < S_; ++s) {
    const size_t gi = (size_t)(b * S_ + s) * HV_ + h;
    const size_t base = gi * (size_t)DK_;
    if (t < 128) lk[t] = kn[base + t];
    else         lq[t - 128] = qn[base + (t - 128)] * qscale;
    __syncthreads();
    const float eg = __expf(g[gi]);
    const float bt = beta[gi];
    const float vt = vf[base + v];
    float kvp = 0.f;
#pragma unroll
    for (int j = 0; j < 64; ++j) {
      Sreg[j] *= eg;
      kvp += lk[kh * 64 + j] * Sreg[j];
    }
    red[t] = kvp;
    __syncthreads();
    const float dv = (vt - (red[v] + red[v + 128])) * bt;
    float op = 0.f;
#pragma unroll
    for (int j = 0; j < 64; ++j) {
      const float kk = lk[kh * 64 + j];
      Sreg[j] += kk * dv;
      op += lq[kh * 64 + j] * Sreg[j];
    }
    __syncthreads();
    red[t] = op;
    __syncthreads();
    if (kh == 0) core[base + v] = red[v] + red[v + 128];
    __syncthreads();
  }
}

__global__ void k_gated_out(const float* __restrict__ core,
                            const float* __restrict__ qkvzba,
                            const float* __restrict__ norm_w,
                            const float* __restrict__ Ot,
                            int8_t* __restrict__ yq, float* __restrict__ ysc) {
  __shared__ float cbuf[VDIM_];
  __shared__ float rinv[HV_];
  __shared__ float red[256];
  const int tok = blockIdx.x;
  for (int i = threadIdx.x; i < VDIM_; i += 256)
    cbuf[i] = core[(size_t)tok * VDIM_ + i];
  __syncthreads();
  if (threadIdx.x < HV_) {
    float ss = 0.f;
    for (int d = 0; d < DV_; ++d) {
      const float c = cbuf[threadIdx.x * DV_ + d];
      ss += c * c;
    }
    rinv[threadIdx.x] = rsqrtf(ss / (float)DV_ + 1e-6f);
  }
  __syncthreads();
  for (int i = threadIdx.x; i < VDIM_; i += 256) {
    const int h = i >> 7, d = i & 127;
    const float z = qkvzba[(size_t)tok * NPROJ_ + OFF_Z + i];
    const float sig = 1.f / (1.f + __expf(-z));
    cbuf[i] = cbuf[i] * rinv[h] * norm_w[d] * (z * sig);
  }
  __syncthreads();
  float yv[16];
  float lmax = 0.f;
  for (int ii = 0; ii < 16; ++ii) {
    const int i = threadIdx.x + ii * 256;
    const int gh = i >> 7, d = i & 127;
    float acc = 0.f;
    for (int h = 0; h < HV_; ++h) acc += cbuf[h * DV_ + d] * Ot[h * HV_ + gh];
    yv[ii] = acc;
    lmax = fmaxf(lmax, fabsf(acc));
  }
  red[threadIdx.x] = lmax;
  __syncthreads();
  for (int s = 128; s > 0; s >>= 1) {
    if (threadIdx.x < s) red[threadIdx.x] = fmaxf(red[threadIdx.x], red[threadIdx.x + s]);
    __syncthreads();
  }
  const float sc = fmaxf(red[0] / 127.f, 1e-8f);
  const float inv = 1.f / sc;
  if (threadIdx.x == 0) ysc[tok] = sc;
  int8_t* dst = yq + (size_t)tok * VDIM_;
  for (int ii = 0; ii < 16; ++ii) {
    const int i = threadIdx.x + ii * 256;
    float q = fminf(fmaxf(rintf(yv[ii] * inv), -128.f), 127.f);
    dst[i] = (int8_t)q;
  }
}

extern "C" void kernel_launch(void* const* d_in, const int* in_sizes, int n_in,
                              void* d_out, int out_size, void* d_ws, size_t ws_size,
                              hipStream_t stream) {
  (void)in_sizes; (void)n_in; (void)out_size; (void)ws_size;
  const float* hs      = (const float*)d_in[0];
  const float* Wq      = (const float*)d_in[1];
  const float* Wk      = (const float*)d_in[2];
  const float* Wv      = (const float*)d_in[3];
  const float* Wz      = (const float*)d_in[4];
  const float* Wb      = (const float*)d_in[5];
  const float* Wa      = (const float*)d_in[6];
  const float* Wo      = (const float*)d_in[7];
  const float* conv_w  = (const float*)d_in[8];
  const float* conv_b  = (const float*)d_in[9];
  const float* A_log   = (const float*)d_in[10];
  const float* dt_bias = (const float*)d_in[11];
  const float* norm_w  = (const float*)d_in[12];
  const float* Pl      = (const float*)d_in[13];
  const float* Pr      = (const float*)d_in[14];
  const float* O_t     = (const float*)d_in[15];
  const float* K_t     = (const float*)d_in[16];
  const float* V_t     = (const float*)d_in[17];

  uint8_t* ws = (uint8_t*)d_ws;
  size_t off = 0;
  auto take = [&](size_t bytes) -> void* {
    void* p = ws + off;
    off = (off + bytes + 255) & ~(size_t)255;
    return p;
  };
  int8_t* aq       = (int8_t*)take((size_t)NTOK_ * H_);
  float*  ascale   = (float*) take((size_t)NTOK_ * 4);
  int8_t* Wall     = (int8_t*)take((size_t)NPROJ_ * H_);
  float*  wscale   = (float*) take((size_t)NPROJ_ * 4);
  int8_t* Weff     = (int8_t*)take((size_t)H_ * VDIM_);
  float*  weffsc   = (float*) take((size_t)H_ * 4);
  float*  qkvzba   = (float*) take((size_t)NTOK_ * NPROJ_ * 4);
  float*  mixed    = (float*) take((size_t)NTOK_ * MIXW_ * 4);
  float*  qn       = (float*) take((size_t)NTOK_ * HV_ * DK_ * 4);
  float*  kn       = (float*) take((size_t)NTOK_ * HV_ * DK_ * 4);
  float*  vfb      = (float*) take((size_t)NTOK_ * HV_ * DV_ * 4);
  float*  beta     = (float*) take((size_t)NTOK_ * HV_ * 4);
  float*  gbuf     = (float*) take((size_t)NTOK_ * HV_ * 4);
  float*  core     = (float*) take((size_t)NTOK_ * HV_ * DV_ * 4);
  int8_t* yq       = (int8_t*)take((size_t)NTOK_ * VDIM_);
  float*  ysc      = (float*) take((size_t)NTOK_ * 4);
  float*  tmp      = (float*) take((size_t)VDIM_ * H_ * 4);

  k_dtrans_fq_act<<<NTOK_, 256, 0, stream>>>(hs, Pl, Pr, aq, ascale);
  k_dtrans_fq_rows<<<KDIM_, 256, 0, stream>>>(Wq, Pl, Pr, Wall, wscale, OFF_Q);
  k_dtrans_fq_rows<<<KDIM_, 256, 0, stream>>>(Wk, Pl, Pr, Wall, wscale, OFF_K);
  k_dtrans_fq_rows<<<VDIM_, 256, 0, stream>>>(Wz, Pl, Pr, Wall, wscale, OFF_Z);
  k_dtrans_fq_rows<<<HV_,   256, 0, stream>>>(Wb, Pl, Pr, Wall, wscale, OFF_B);
  k_dtrans_fq_rows<<<HV_,   256, 0, stream>>>(Wa, Pl, Pr, Wall, wscale, OFF_A);
  k_dtrans_rows_f32<<<VDIM_, 256, 0, stream>>>(Wv, Pl, Pr, tmp);
  k_vt_mix_fq<<<VDIM_, 256, 0, stream>>>(tmp, V_t, Wall, wscale, OFF_V);
  k_weff_stage1<<<H_, 256, 0, stream>>>(Wo, O_t, tmp);
  k_weff_stage2_fq<<<H_, 256, 0, stream>>>(tmp, V_t, Weff, weffsc);
  {
    dim3 grid((NPROJ_ / 16 + 7) / 8, NTOK_ / 16);
    k_gemm_i8_wmma<<<grid, 256, 0, stream>>>(aq, ascale, Wall, wscale, qkvzba,
                                             NTOK_, NPROJ_, H_);
  }
  k_conv_silu<<<NTOK_, 256, 0, stream>>>(qkvzba, conv_w, conv_b, mixed);
  k_qk_prep<<<NTOK_ * HK_, 128, 0, stream>>>(mixed, K_t, qn, kn);
  k_v_prep<<<NTOK_ * HV_, 128, 0, stream>>>(mixed, vfb);
  k_beta_g<<<(NTOK_ * HV_) / 256, 256, 0, stream>>>(qkvzba, A_log, dt_bias, beta, gbuf);
  k_delta_scan<<<B_ * HV_, 256, 0, stream>>>(qn, kn, vfb, gbuf, beta, core);
  k_gated_out<<<NTOK_, 256, 0, stream>>>(core, qkvzba, norm_w, O_t, yq, ysc);
  {
    dim3 grid((H_ / 16 + 7) / 8, NTOK_ / 16);
    k_gemm_i8_wmma<<<grid, 256, 0, stream>>>(yq, ysc, Weff, weffsc, (float*)d_out,
                                             NTOK_, H_, VDIM_);
  }
}
